// LlamaForCFG_49572512531165
// MI455X (gfx1250) — compile-verified, matrix operand order checked by reading, not run on hardware
//
#include <hip/hip_runtime.h>
#include <hip/hip_bf16.h>

// ---------------------------------------------------------------------------
// Model constants (from reference)
// ---------------------------------------------------------------------------
#define NL   30
#define NH   16
#define HD   64
#define DM   1024
#define DFF  4096
#define VOC  8192
#define NB   2
#define SQ   512
#define PP   128
#define TT   640          // PP + SQ

typedef __bf16 v16bf __attribute__((ext_vector_type(16)));
typedef __bf16 bf4   __attribute__((ext_vector_type(4)));
typedef float  v8f   __attribute__((ext_vector_type(8)));

static __device__ inline bf4 cvt4(float4 f)
{
    bf4 r;
    r[0] = (__bf16)f.x; r[1] = (__bf16)f.y;
    r[2] = (__bf16)f.z; r[3] = (__bf16)f.w;
    return r;
}

// Workgroup barrier that only drains the LDS counter (dscnt) before the
// split-barrier signal/wait pair. Unlike __syncthreads() (acq_rel fence ->
// s_wait_loadcnt_dscnt 0x0), this leaves prefetched *global* loads in flight
// across the barrier: they target registers, so no other wave can observe
// them, and the fragment ds_loads were already drained by the s_wait_dscnt
// the compiler inserts before the consuming WMMAs.
static __device__ inline void wg_barrier_lds()
{
    asm volatile("s_wait_dscnt 0x0\n\t"
                 "s_barrier_signal -1\n\t"
                 "s_barrier_wait -1" ::: "memory");
}

// ---------------------------------------------------------------------------
// RMSNorm: one block per row
// ---------------------------------------------------------------------------
__global__ __launch_bounds__(256) void rmsnorm_kernel(const float* __restrict__ x,
                                                      const float* __restrict__ w,
                                                      float* __restrict__ out, int dim)
{
    __shared__ float red[256];
    long row = blockIdx.x;
    const float* xr = x + row * dim;
    float ss = 0.f;
    for (int i = threadIdx.x; i < dim; i += 256) { float v = xr[i]; ss += v * v; }
    red[threadIdx.x] = ss; __syncthreads();
    for (int s = 128; s > 0; s >>= 1) {
        if (threadIdx.x < s) red[threadIdx.x] += red[threadIdx.x + s];
        __syncthreads();
    }
    float inv = rsqrtf(red[0] / (float)dim + 1e-5f);
    float* orow = out + row * dim;
    for (int i = threadIdx.x; i < dim; i += 256) orow[i] = xr[i] * inv * w[i];
}

// ---------------------------------------------------------------------------
// Generic batched WMMA GEMM:  Out = A(MxK) * B(KxN) [+ Res], bf16 compute.
//
// Block: 256 threads = 8 waves; block tile 64(M) x 64(N); each wave computes
// two 16x16 tiles (m, m+16) sharing one B fragment -> 2 WMMA per k-step.
// K-step 32. Tiles staged to ping-pong LDS buffers in *fragment-major*
// permuted layout (p = 16*((k>>3)&1) + (k&7) + 8*(k>>4)) so each lane's WMMA
// operand is 32 contiguous bytes -> ds_load_b128 pairs.
//
// Two-tile-deep software pipeline: tile j is loaded into register set (j&1)
// and staged into LDS buffer (j&1). The main loop is unrolled by 2 so the
// register-set selection is compile-time; global loads for tile kt+2 are
// issued a full iteration before their s_wait/convert/ds_store, hiding HBM
// latency under the WMMAs. One dscnt-only split barrier per k-step keeps
// those prefetch loads in flight across the barrier.
//
// BT (B-transposed) template parameter: BT=1 reads B as N x K (QK^T),
// BT=0 reads B as K x N (weights, PV).
// ---------------------------------------------------------------------------
template <int BT>
__global__ __launch_bounds__(256) void gemm_wmma_kernel(
    const float* __restrict__ A, const float* __restrict__ Bm,
    const float* __restrict__ Res, float* __restrict__ Out,
    int K, int lda, int ldb, int ldc,
    long long sAb, long long sAh, long long sBb, long long sBh,
    long long sOb, long long sOh, int nH)
{
    __shared__ alignas(32) __bf16 As[2][64][48];   // row stride 96B (32B aligned)
    __shared__ alignas(32) __bf16 Bs[2][64][48];   // fragment-major, [n][p]

    int z  = blockIdx.z;
    int bi = z / nH, hi = z % nH;
    A   += (long long)bi * sAb + (long long)hi * sAh;
    Bm  += (long long)bi * sBb + (long long)hi * sBh;
    Out += (long long)bi * sOb + (long long)hi * sOh;
    if (Res) Res += (long long)bi * sOb + (long long)hi * sOh;

    const int m0   = blockIdx.y * 64;
    const int n0   = blockIdx.x * 64;
    const int tid  = threadIdx.x;
    const int lane = tid & 31;
    const int wave = tid >> 5;
    const int mw   = (wave & 1) * 32;      // wave m base (covers mw, mw+16)
    const int nw   = (wave >> 1) * 16;     // wave n base

    // Staging indices: A (and transposed B) as 64 rows x 32 k, float4 chunks.
    const int ar = tid >> 3;               // 0..31; rows ar and ar+32
    const int ak = (tid & 7) * 4;          // k column within tile
    // Non-transposed B: 32 k x 64 n, float4 over n.
    const int kb = tid >> 4;               // 0..15; k rows kb and kb+16
    const int nb = (tid & 15) * 4;         // n column within tile

    // fragment-major permutation p(k)
    const int pa  = ((ak >> 3) & 1) * 16 + (ak & 7) + ((ak >> 4) << 3);
    const int pb0 = ((kb >> 3) & 1) * 16 + (kb & 7) + ((kb >> 4) << 3);
    const int kb1 = kb + 16;
    const int pb1 = ((kb1 >> 3) & 1) * 16 + (kb1 & 7) + ((kb1 >> 4) << 3);

    // Two register sets: r[0..1]=A rows, r[2..3]=B
    float4 rA[4], rB[4];

    auto loadTiles = [&](int kt, float4* r) {
        const int k0 = kt << 5;
        r[0] = *(const float4*)&A[(long long)(m0 + ar) * lda + (k0 + ak)];
        r[1] = *(const float4*)&A[(long long)(m0 + ar + 32) * lda + (k0 + ak)];
        if (BT) {
            r[2] = *(const float4*)&Bm[(long long)(n0 + ar) * ldb + (k0 + ak)];
            r[3] = *(const float4*)&Bm[(long long)(n0 + ar + 32) * ldb + (k0 + ak)];
        } else {
            r[2] = *(const float4*)&Bm[(long long)(k0 + kb) * ldb + (n0 + nb)];
            r[3] = *(const float4*)&Bm[(long long)(k0 + kb1) * ldb + (n0 + nb)];
        }
    };
    auto storeTiles = [&](int buf, const float4* r) {
        *(bf4*)&As[buf][ar][pa]      = cvt4(r[0]);
        *(bf4*)&As[buf][ar + 32][pa] = cvt4(r[1]);
        if (BT) {
            *(bf4*)&Bs[buf][ar][pa]      = cvt4(r[2]);
            *(bf4*)&Bs[buf][ar + 32][pa] = cvt4(r[3]);
        } else {
            Bs[buf][nb + 0][pb0] = (__bf16)r[2].x; Bs[buf][nb + 1][pb0] = (__bf16)r[2].y;
            Bs[buf][nb + 2][pb0] = (__bf16)r[2].z; Bs[buf][nb + 3][pb0] = (__bf16)r[2].w;
            Bs[buf][nb + 0][pb1] = (__bf16)r[3].x; Bs[buf][nb + 1][pb1] = (__bf16)r[3].y;
            Bs[buf][nb + 2][pb1] = (__bf16)r[3].z; Bs[buf][nb + 3][pb1] = (__bf16)r[3].w;
        }
    };

    v8f acc0 = {}, acc1 = {};
    const int grp  = (lane >> 4) << 4;     // 0 or 16 elements
    const int l15  = lane & 15;

    auto compute = [&](int buf) {
        v16bf af0 = *(const v16bf*)&As[buf][mw + l15][grp];
        v16bf af1 = *(const v16bf*)&As[buf][mw + 16 + l15][grp];
        v16bf bfm = *(const v16bf*)&Bs[buf][nw + l15][grp];
        acc0 = __builtin_amdgcn_wmma_f32_16x16x32_bf16(false, af0, false, bfm,
                                                       (short)0, acc0, false, false);
        acc1 = __builtin_amdgcn_wmma_f32_16x16x32_bf16(false, af1, false, bfm,
                                                       (short)0, acc1, false, false);
    };

    const int nk = K >> 5;                 // number of 32-wide k tiles (>= 1)

    // Pipeline prologue: tile0 -> setA/buf0; tile1 loads in flight in setB.
    loadTiles(0, rA);
    if (nk > 1) loadTiles(1, rB);
    storeTiles(0, rA);
    wg_barrier_lds();

    for (int kt = 0; kt < nk; kt += 2) {
        // ---- even step: tile kt lives in buf0 ----
        compute(0);
        if (kt + 2 < nk) loadTiles(kt + 2, rA);   // issue early: consumed next step
        if (kt + 1 < nk) storeTiles(1, rB);       // tile kt+1 (loaded last step)
        wg_barrier_lds();

        if (kt + 1 < nk) {
            // ---- odd step: tile kt+1 lives in buf1 ----
            compute(1);
            if (kt + 3 < nk) loadTiles(kt + 3, rB);
            if (kt + 2 < nk) storeTiles(0, rA);   // tile kt+2
            wg_barrier_lds();
        }
    }

    float* a0p = (float*)&acc0;
    float* a1p = (float*)&acc1;
    int n = n0 + nw + l15;
#pragma unroll
    for (int r = 0; r < 8; ++r) {
        int m = m0 + mw + r + ((lane >> 4) << 3);
        long long i0 = (long long)m * ldc + n;
        long long i1 = (long long)(m + 16) * ldc + n;
        float v0 = a0p[r], v1 = a1p[r];
        if (Res) { v0 += Res[i0]; v1 += Res[i1]; }
        Out[i0] = v0;
        Out[i1] = v1;
    }
}

// ---------------------------------------------------------------------------
// Masked softmax over score rows (applies SCALE, key mask, causal mask)
// One block per (b,h,q) row of length TT.
// ---------------------------------------------------------------------------
__global__ __launch_bounds__(256) void softmax_mask_kernel(float* __restrict__ sc,
                                                           const int* __restrict__ amask)
{
    __shared__ float red[256];
    long row = blockIdx.x;                 // over NB*NH*SQ
    int q = (int)(row % SQ);
    int b = (int)(row / ((long)NH * SQ));
    float* r = sc + row * TT;
    int limit = PP + q;                    // allowed: t <= limit
    const float scale = 0.125f;            // 1/sqrt(64)
    const float NEGF  = -3.402823466e38f;

    float vals[3];
    float mx = NEGF;
    for (int t = threadIdx.x, i = 0; t < TT; t += 256, ++i) {
        float km = (1.f - (float)amask[b * TT + t]) * NEGF;
        float v  = (t <= limit) ? (r[t] * scale + km) : NEGF;
        vals[i] = v;
        mx = fmaxf(mx, v);
    }
    red[threadIdx.x] = mx; __syncthreads();
    for (int s = 128; s > 0; s >>= 1) {
        if (threadIdx.x < s) red[threadIdx.x] = fmaxf(red[threadIdx.x], red[threadIdx.x + s]);
        __syncthreads();
    }
    mx = red[0]; __syncthreads();

    float sum = 0.f;
    for (int t = threadIdx.x, i = 0; t < TT; t += 256, ++i) {
        float e = __expf(vals[i] - mx);
        vals[i] = e;
        sum += e;
    }
    red[threadIdx.x] = sum; __syncthreads();
    for (int s = 128; s > 0; s >>= 1) {
        if (threadIdx.x < s) red[threadIdx.x] += red[threadIdx.x + s];
        __syncthreads();
    }
    float inv = 1.f / red[0];
    for (int t = threadIdx.x, i = 0; t < TT; t += 256, ++i) r[t] = vals[i] * inv;
}

// ---------------------------------------------------------------------------
// RoPE applied in place to q and k_new ([B*S, DM] layout); one thread per
// (b,s,h,j) pair rotating (j, j+32).
// ---------------------------------------------------------------------------
__global__ __launch_bounds__(256) void rope_kernel(float* __restrict__ q, float* __restrict__ k)
{
    long idx = (long)blockIdx.x * 256 + threadIdx.x;   // NB*SQ*NH*32
    int j = idx & 31;
    long t = idx >> 5;
    int h = (int)(t & 15); t >>= 4;
    int s = (int)(t & 511);
    int b = (int)(t >> 9);

    float pos = (float)(PP + s);
    float inv = __powf(10000.f, -(float)(2 * j) / 64.f);
    float ang = pos * inv;
    float c = __cosf(ang), sn = __sinf(ang);

    long base = (((long)b * SQ + s) * DM) + h * HD;
    float q0 = q[base + j], q1 = q[base + j + 32];
    q[base + j]      = q0 * c - q1 * sn;
    q[base + j + 32] = q1 * c + q0 * sn;
    float k0 = k[base + j], k1 = k[base + j + 32];
    k[base + j]      = k0 * c - k1 * sn;
    k[base + j + 32] = k1 * c + k0 * sn;
}

// ---------------------------------------------------------------------------
// Build full K/V cache for this layer directly in the pks/pvs output regions:
// layout [B,H,T,D]; t<P from past cache, t>=P from the new projections.
// ---------------------------------------------------------------------------
__global__ __launch_bounds__(256) void kvfill_kernel(const float* __restrict__ pastK,
                                                     const float* __restrict__ pastV,
                                                     const float* __restrict__ kx,
                                                     const float* __restrict__ vx,
                                                     float* __restrict__ outK,
                                                     float* __restrict__ outV)
{
    long idx = (long)blockIdx.x * 256 + threadIdx.x;   // NB*NH*TT*HD
    int d = idx & 63;
    long t2 = idx >> 6;
    int t = (int)(t2 % TT);
    long t3 = t2 / TT;
    int h = (int)(t3 & 15);
    int b = (int)(t3 >> 4);

    float kv, vv;
    if (t < PP) {
        long pi = ((((long)b * NH + h) * PP + t) * HD) + d;
        kv = pastK[pi]; vv = pastV[pi];
    } else {
        int s = t - PP;
        long si = (((long)b * SQ + s) * DM) + h * HD + d;
        kv = kx[si]; vv = vx[si];
    }
    outK[idx] = kv;
    outV[idx] = vv;
}

// ---------------------------------------------------------------------------
// gate = silu(gate) * up
// ---------------------------------------------------------------------------
__global__ __launch_bounds__(256) void silu_mul_kernel(float* __restrict__ gate,
                                                       const float* __restrict__ up)
{
    long i = (long)blockIdx.x * 256 + threadIdx.x;
    float g = gate[i];
    gate[i] = (g / (1.f + __expf(-g))) * up[i];
}

// ---------------------------------------------------------------------------
// CFG combine: logits = cond + w * (cond - uncond)
// ---------------------------------------------------------------------------
__global__ __launch_bounds__(256) void cfg_kernel(const float* __restrict__ lf,
                                                  const float* __restrict__ w,
                                                  float* __restrict__ out)
{
    long i = (long)blockIdx.x * 256 + threadIdx.x;     // SQ*VOC
    float c = lf[i];
    float u = lf[i + (long)SQ * VOC];
    out[i] = c + w[0] * (c - u);
}

// ---------------------------------------------------------------------------
// Host-side GEMM launcher (block tile 64x64)
// ---------------------------------------------------------------------------
static inline void gemm(hipStream_t st, const float* A, const float* Bm,
                        const float* Res, float* Out,
                        int M, int N, int K, int lda, int ldb, int ldc,
                        long long sAb, long long sAh, long long sBb, long long sBh,
                        long long sOb, long long sOh, int nH, int nBat, int bTrans)
{
    dim3 g(N / 64, M / 64, nBat);
    if (bTrans)
        gemm_wmma_kernel<1><<<g, 256, 0, st>>>(A, Bm, Res, Out, K, lda, ldb, ldc,
                                               sAb, sAh, sBb, sBh, sOb, sOh, nH);
    else
        gemm_wmma_kernel<0><<<g, 256, 0, st>>>(A, Bm, Res, Out, K, lda, ldb, ldc,
                                               sAb, sAh, sBb, sBh, sOb, sOh, nH);
}

extern "C" void kernel_launch(void* const* d_in, const int* in_sizes, int n_in,
                              void* d_out, int out_size, void* d_ws, size_t ws_size,
                              hipStream_t stream)
{
    (void)in_sizes; (void)n_in; (void)out_size; (void)ws_size;

    const float* inputs_embeds = (const float*)d_in[0];
    const int*   attn_mask     = (const int*)  d_in[1];
    const float* cfg_w         = (const float*)d_in[2];
    const float* past_k        = (const float*)d_in[3];
    const float* past_v        = (const float*)d_in[4];
    const float* Wq            = (const float*)d_in[5];
    const float* Wk            = (const float*)d_in[6];
    const float* Wv            = (const float*)d_in[7];
    const float* Wo            = (const float*)d_in[8];
    const float* Wgate         = (const float*)d_in[9];
    const float* Wup           = (const float*)d_in[10];
    const float* Wdown         = (const float*)d_in[11];
    const float* ln1           = (const float*)d_in[12];
    const float* ln2           = (const float*)d_in[13];
    const float* norm_w        = (const float*)d_in[14];
    const float* Whead         = (const float*)d_in[15];

    // Workspace layout (floats), U = 1Mi elements
    const long long U = 1048576;
    float* ws   = (float*)d_ws;
    float* x    = ws;            // [B*S, DM]            1U
    float* h    = ws +  1 * U;   // [B*S, DM]            1U
    float* q    = ws +  2 * U;   // [B*S, DM]            1U
    float* kx   = ws +  3 * U;   // [B*S, DM]            1U
    float* vx   = ws +  4 * U;   // [B*S, DM]            1U
    float* ao   = ws +  5 * U;   // [B*S, DM]            1U
    float* gate = ws +  6 * U;   // [B*S, DFF]           4U
    float* up   = ws + 10 * U;   // [B*S, DFF]           4U
    float* sc   = ws + 14 * U;   // [B,H,S,T]           10U
    float* lf   = ws + 24 * U;   // [B*S, VOC]           8U

    // Output layout (floats): logits | caps | pks | pvs
    float* o      = (float*)d_out;
    float* o_log  = o;                                       // SQ*VOC
    float* o_caps = o_log  + (long long)SQ * VOC;            // 4*NH*SQ*TT
    float* o_pks  = o_caps + 4LL * NH * SQ * TT;             // NL*NB*NH*TT*HD
    float* o_pvs  = o_pks  + (long long)NL * NB * NH * TT * HD;

    const int MROWS = NB * SQ;   // 1024

    // x = inputs_embeds
    hipMemcpyAsync(x, inputs_embeds, (size_t)MROWS * DM * sizeof(float),
                   hipMemcpyDeviceToDevice, stream);

    // attention batched-GEMM strides
    const long long saB = (long long)SQ * DM, saH = HD;          // q / ao in [B*S, DM]
    const long long skB = (long long)NH * TT * HD, skH = (long long)TT * HD;  // KV cache [B,H,T,D]
    const long long ssB = (long long)NH * SQ * TT, ssH = (long long)SQ * TT;  // scores [B,H,S,T]

    int capIdx = 0;
    for (int i = 0; i < NL; ++i) {
        const float* wq = Wq + (long long)i * DM * DM;
        const float* wk = Wk + (long long)i * DM * DM;
        const float* wv = Wv + (long long)i * DM * DM;
        const float* wo = Wo + (long long)i * DM * DM;
        const float* wg = Wgate + (long long)i * DM * DFF;
        const float* wu = Wup   + (long long)i * DM * DFF;
        const float* wd = Wdown + (long long)i * DFF * DM;

        // h = rmsnorm(x, ln1[i])
        rmsnorm_kernel<<<MROWS, 256, 0, stream>>>(x, ln1 + (long long)i * DM, h, DM);

        // q/k/v projections
        gemm(stream, h, wq, nullptr, q,  MROWS, DM, DM, DM, DM, DM, 0,0,0,0,0,0, 1, 1, 0);
        gemm(stream, h, wk, nullptr, kx, MROWS, DM, DM, DM, DM, DM, 0,0,0,0,0,0, 1, 1, 0);
        gemm(stream, h, wv, nullptr, vx, MROWS, DM, DM, DM, DM, DM, 0,0,0,0,0,0, 1, 1, 0);

        // RoPE on q and k_new
        rope_kernel<<<(NB * SQ * NH * 32) / 256, 256, 0, stream>>>(q, kx);

        // Build this layer's full K/V cache directly into pks/pvs outputs
        float* Kc = o_pks + (long long)i * NB * NH * TT * HD;
        float* Vc = o_pvs + (long long)i * NB * NH * TT * HD;
        kvfill_kernel<<<(NB * NH * TT * HD) / 256, 256, 0, stream>>>(
            past_k + (long long)i * NB * NH * PP * HD,
            past_v + (long long)i * NB * NH * PP * HD, kx, vx, Kc, Vc);

        // scores[b,h] = q[b,h] (SxD) @ K[b,h]^T (DxT)
        gemm(stream, q, Kc, nullptr, sc, SQ, TT, HD, DM, HD, TT,
             saB, saH, skB, skH, ssB, ssH, NH, NB * NH, 1);

        // softmax with scale + masks
        softmax_mask_kernel<<<NB * NH * SQ, 256, 0, stream>>>(sc, attn_mask);

        // capture attention weights for batch 0 on ALIGN layers
        if (i == 9 || i == 12 || i == 13) {
            hipMemcpyAsync(o_caps + (long long)capIdx * NH * SQ * TT, sc,
                           (size_t)NH * SQ * TT * sizeof(float),
                           hipMemcpyDeviceToDevice, stream);
            ++capIdx;
        }

        // ao[b,h] = probs (SxT) @ V[b,h] (TxD)
        gemm(stream, sc, Vc, nullptr, ao, SQ, HD, TT, TT, HD, DM,
             ssB, ssH, skB, skH, saB, saH, NH, NB * NH, 0);

        // x = x + ao @ Wo[i]
        gemm(stream, ao, wo, x, x, MROWS, DM, DM, DM, DM, DM, 0,0,0,0,0,0, 1, 1, 0);

        // MLP
        rmsnorm_kernel<<<MROWS, 256, 0, stream>>>(x, ln2 + (long long)i * DM, h, DM);
        gemm(stream, h, wg, nullptr, gate, MROWS, DFF, DM, DM, DFF, DFF, 0,0,0,0,0,0, 1, 1, 0);
        gemm(stream, h, wu, nullptr, up,   MROWS, DFF, DM, DM, DFF, DFF, 0,0,0,0,0,0, 1, 1, 0);
        silu_mul_kernel<<<(MROWS * DFF) / 256, 256, 0, stream>>>(gate, up);
        gemm(stream, gate, wd, x, x, MROWS, DM, DFF, DFF, DM, DM, 0,0,0,0,0,0, 1, 1, 0);
    }

    // Final norm + LM head + CFG combine
    rmsnorm_kernel<<<MROWS, 256, 0, stream>>>(x, norm_w, h, DM);
    gemm(stream, h, Whead, nullptr, lf, MROWS, VOC, DM, DM, VOC, VOC, 0,0,0,0,0,0, 1, 1, 0);
    cfg_kernel<<<((long long)SQ * VOC) / 256, 256, 0, stream>>>(lf, cfg_w, o_log);
}
